// DenseAttentionOneHead_34600256536774
// MI455X (gfx1250) — matrix operand order repaired and checked by reading, compiled, not run on hardware
//
#include <hip/hip_runtime.h>
#include <hip/hip_bf16.h>

// attention = (x W^T)(x^T x) = x * (W^T * (x^T x))   -- exact reassociation,
// turns an O(S^2 H) problem (309 GFLOP + 256MB intermediate) into
// O(S H^2) (77 GFLOP, only H x H intermediates). All GEMMs run on
// v_wmma_f32_16x16x32_bf16 (bf16 operands, f32 accumulate).
//
// v4: 128x128 block tile (8 waves x 32x64 wave tiles, 8 WMMA/wave/step),
//     ping-pong LDS double buffering with register staging (1 barrier/step),
//     global_prefetch two K-steps ahead; transposed staging packs k-pairs
//     with single v_perm_b32 ops feeding ds_store_b32/_2addr_b32.

typedef __bf16 bf16;
typedef __attribute__((ext_vector_type(16))) __bf16 v16bf;
typedef __attribute__((ext_vector_type(8)))  __bf16 v8bf;
typedef __attribute__((ext_vector_type(4)))  __bf16 v4bf;
typedef __attribute__((ext_vector_type(8)))  float  v8f;

#define BM 128
#define BN 128
#define BK 32
#define BKP 36   // padded LDS K-stride (72B rows; conflict-mitigating, 8B aligned)

union Frag16 { v16bf v; v4bf q[4]; };
union Chunk8 { v8bf v; v4bf h[2]; unsigned d[4]; };

// byte-select packing of two bf16 rows into one dword column-pair
#define SEL_LO 0x05040100u   // {hi.b1,hi.b0, lo.b1,lo.b0} -> even element
#define SEL_HI 0x07060302u   // {hi.b3,hi.b2, lo.b3,lo.b2} -> odd element

// ---------------------------------------------------------------------------
// fp32 -> bf16 cast (grid-stride)
// ---------------------------------------------------------------------------
__global__ __launch_bounds__(256)
void cast_f32_bf16(const float* __restrict__ in, bf16* __restrict__ out, long long n) {
    long long i = (long long)blockIdx.x * blockDim.x + threadIdx.x;
    long long stride = (long long)gridDim.x * blockDim.x;
    for (; i < n; i += stride) out[i] = (bf16)in[i];
}

// ---------------------------------------------------------------------------
// Batched GEMM: C[M,N] = opA(A) @ B[K,N]
//   AT=false: A is [M,K] row-major (lda = K-stride)
//   AT=true : A is stored [K,M] row-major; logical A[m,k] = Ast[k*lda + m]
// B always [K,N] row-major. C row-major, OutT = float or bf16.
// All of M,N,K are multiples of the tile sizes for our problem -- no bounds.
// ---------------------------------------------------------------------------
template<bool AT, typename OutT>
__global__ __launch_bounds__(256)
void gemm_bf16_wmma(const bf16* __restrict__ Ag, const bf16* __restrict__ Bg,
                    OutT* __restrict__ Cg,
                    int M, int N, int K, int lda, int ldb, int ldc,
                    long long strideA, long long strideB, long long strideC)
{
    const int tid  = threadIdx.x;
    const int bz   = blockIdx.z;
    const bf16* A = Ag + (long long)bz * strideA;
    const bf16* B = Bg + (long long)bz * strideB;
    OutT*       C = Cg + (long long)bz * strideC;

    const int m0 = blockIdx.y * BM;
    const int n0 = blockIdx.x * BN;

    __shared__ bf16 As[2][BM][BKP];
    __shared__ bf16 Bs[2][BN][BKP];   // stored transposed: Bs[buf][n][k]

    const int w     = tid >> 5;        // wave 0..7
    const int lane  = tid & 31;
    const int wm    = (w & 3) * 32;    // wave tile M offset (4 waves along M)
    const int wn    = (w >> 2) * 64;   // wave tile N offset (2 waves along N)
    const int lrow  = lane & 15;
    const int khalf = lane >> 4;       // 0: lanes 0-15, 1: lanes 16-31

    // --- per-thread staging coordinates ---
    // A, !AT: row-major [BM][BK]: 512 chunks of 8, 4 per row (2 per thread)
    const int aRow  = tid >> 2,  aKc = (tid & 3)  << 3;   // + it*64 rows
    // A, AT : memory tile is [BK][BM]; thread owns k-pair rows {2kp, 2kp+1} x 8 cols
    const int aKp   = tid >> 4,  aMc = (tid & 15) << 3;
    // B     : memory tile is [BK][BN]; thread owns k-pair rows {2kp, 2kp+1} x 8 cols
    const int bKp   = tid >> 4,  bNc = (tid & 15) << 3;

    v8f acc[2][4] = {};

    Chunk8 ra[2], rb[2];   // AT/B: ra/rb[0] = row 2kp, ra/rb[1] = row 2kp+1

    // ---- helper lambdas -------------------------------------------------
    auto load_tile_regs = [&](int kt) {
        if (!AT) {
            #pragma unroll
            for (int it = 0; it < 2; ++it)
                ra[it].v = *reinterpret_cast<const v8bf*>(
                    A + (long long)(m0 + aRow + it * 64) * lda + kt + aKc);
        } else {
            #pragma unroll
            for (int it = 0; it < 2; ++it)
                ra[it].v = *reinterpret_cast<const v8bf*>(
                    A + (long long)(kt + 2 * aKp + it) * lda + m0 + aMc);
        }
        #pragma unroll
        for (int it = 0; it < 2; ++it)
            rb[it].v = *reinterpret_cast<const v8bf*>(
                B + (long long)(kt + 2 * bKp + it) * ldb + n0 + bNc);
    };
    auto store_tile_lds = [&](int p) {
        if (!AT) {
            #pragma unroll
            for (int it = 0; it < 2; ++it) {
                *reinterpret_cast<v4bf*>(&As[p][aRow + it * 64][aKc])     = ra[it].h[0];
                *reinterpret_cast<v4bf*>(&As[p][aRow + it * 64][aKc + 4]) = ra[it].h[1];
            }
        } else {
            // pack {A[2kp][m], A[2kp+1][m]} -> one dword at As[m][2kp] via v_perm_b32
            #pragma unroll
            for (int i = 0; i < 4; ++i) {
                unsigned lo = __builtin_amdgcn_perm(ra[1].d[i], ra[0].d[i], SEL_LO);
                unsigned hi = __builtin_amdgcn_perm(ra[1].d[i], ra[0].d[i], SEL_HI);
                *reinterpret_cast<unsigned*>(&As[p][aMc + 2 * i][2 * aKp])     = lo;
                *reinterpret_cast<unsigned*>(&As[p][aMc + 2 * i + 1][2 * aKp]) = hi;
            }
        }
        #pragma unroll
        for (int i = 0; i < 4; ++i) {
            unsigned lo = __builtin_amdgcn_perm(rb[1].d[i], rb[0].d[i], SEL_LO);
            unsigned hi = __builtin_amdgcn_perm(rb[1].d[i], rb[0].d[i], SEL_HI);
            *reinterpret_cast<unsigned*>(&Bs[p][bNc + 2 * i][2 * bKp])     = lo;
            *reinterpret_cast<unsigned*>(&Bs[p][bNc + 2 * i + 1][2 * bKp]) = hi;
        }
    };
    auto prefetch_tile = [&](int kt) {
        if (kt < K) {
            if (!AT)
                __builtin_prefetch(A + (long long)(m0 + aRow) * lda + kt + aKc, 0, 1);
            else
                __builtin_prefetch(A + (long long)(kt + 2 * aKp) * lda + m0 + aMc, 0, 1);
            __builtin_prefetch(B + (long long)(kt + 2 * bKp) * ldb + n0 + bNc, 0, 1);
        }
    };

    // ---- prologue: stage first tile ----
    load_tile_regs(0);
    prefetch_tile(BK);
    store_tile_lds(0);
    __syncthreads();

    int p = 0;
    for (int kt = 0; kt < K; kt += BK) {
        const bool more = (kt + BK) < K;
        if (more) {
            load_tile_regs(kt + BK);
            prefetch_tile(kt + 2 * BK);
        }

        // ---- per-wave fragments (ISA 7.12.2 layouts) ----
        Frag16 af[2], bfr[4];
        #pragma unroll
        for (int rbk = 0; rbk < 2; ++rbk) {
            // A 16x32: lane = row; khalf 0 -> K 0-7 & 16-23, khalf 1 -> K 8-15 & 24-31
            const bf16* ap = &As[p][wm + rbk * 16 + lrow][khalf * 8];
            af[rbk].q[0] = *reinterpret_cast<const v4bf*>(ap);
            af[rbk].q[1] = *reinterpret_cast<const v4bf*>(ap + 4);
            af[rbk].q[2] = *reinterpret_cast<const v4bf*>(ap + 16);
            af[rbk].q[3] = *reinterpret_cast<const v4bf*>(ap + 20);
        }
        #pragma unroll
        for (int cb = 0; cb < 4; ++cb) {
            // B 32x16: lane = col; khalf 0 -> K 0-7 & 16-23, khalf 1 -> K 8-15 & 24-31
            const bf16* bp = &Bs[p][wn + cb * 16 + lrow][khalf * 8];
            bfr[cb].q[0] = *reinterpret_cast<const v4bf*>(bp);
            bfr[cb].q[1] = *reinterpret_cast<const v4bf*>(bp + 4);
            bfr[cb].q[2] = *reinterpret_cast<const v4bf*>(bp + 16);
            bfr[cb].q[3] = *reinterpret_cast<const v4bf*>(bp + 20);
        }

        #pragma unroll
        for (int rbk = 0; rbk < 2; ++rbk)
            #pragma unroll
            for (int cb = 0; cb < 4; ++cb)
                acc[rbk][cb] = __builtin_amdgcn_wmma_f32_16x16x32_bf16(
                    false, af[rbk].v, false, bfr[cb].v,
                    (short)0, acc[rbk][cb], false, false);

        if (more) store_tile_lds(p ^ 1);
        __syncthreads();
        p ^= 1;
    }

    // ---- epilogue: C 16x16 f32 layout: row = 8*khalf + r, col = lane&15 ----
    #pragma unroll
    for (int rbk = 0; rbk < 2; ++rbk) {
        #pragma unroll
        for (int cb = 0; cb < 4; ++cb) {
            const int col = n0 + wn + cb * 16 + lrow;
            #pragma unroll
            for (int r = 0; r < 8; ++r) {
                const int row = m0 + wm + rbk * 16 + khalf * 8 + r;
                C[(long long)row * ldc + col] = (OutT)acc[rbk][cb][r];
            }
        }
    }
    (void)M; (void)N;
}

// ---------------------------------------------------------------------------
extern "C" void kernel_launch(void* const* d_in, const int* in_sizes, int n_in,
                              void* d_out, int out_size, void* d_ws, size_t ws_size,
                              hipStream_t stream) {
    (void)in_sizes; (void)n_in; (void)out_size; (void)ws_size;

    const int Bb = 4, S = 4096, H = 1024;
    const float* x = (const float*)d_in[0];   // [B,S,H] fp32
    const float* W = (const float*)d_in[1];   // [H,H]  fp32
    float* out = (float*)d_out;               // [B,S,H] fp32

    // workspace layout (bf16): xb | Wb | G | Mw   = 32MB + 2MB + 8MB + 8MB
    bf16* xb = (bf16*)d_ws;                       // B*S*H
    bf16* Wb = xb + (long long)Bb * S * H;        // H*H
    bf16* G  = Wb + (long long)H * H;             // B*H*H
    bf16* Mw = G  + (long long)Bb * H * H;        // B*H*H

    const long long nx = (long long)Bb * S * H;
    const long long nw = (long long)H * H;

    cast_f32_bf16<<<8192, 256, 0, stream>>>(x, xb, nx);
    cast_f32_bf16<<<1024, 256, 0, stream>>>(W, Wb, nw);

    // G[b] = x[b]^T @ x[b]          (M=H, N=H, K=S), A = xb with AT
    {
        dim3 grid(H / BN, H / BM, Bb);
        gemm_bf16_wmma<true, bf16><<<grid, 256, 0, stream>>>(
            xb, xb, G, H, H, S, H, H, H,
            (long long)S * H, (long long)S * H, (long long)H * H);
    }
    // Mw[b] = W^T @ G[b]            (M=H, N=H, K=H), A = Wb with AT, batch-shared A
    {
        dim3 grid(H / BN, H / BM, Bb);
        gemm_bf16_wmma<true, bf16><<<grid, 256, 0, stream>>>(
            Wb, G, Mw, H, H, H, H, H, H,
            0LL, (long long)H * H, (long long)H * H);
    }
    // out[b] = x[b] @ Mw[b]         (M=S, N=H, K=H), fp32 output
    {
        dim3 grid(H / BN, S / BM, Bb);
        gemm_bf16_wmma<false, float><<<grid, 256, 0, stream>>>(
            xb, Mw, out, S, H, H, H, H, H,
            (long long)S * H, (long long)H * H, (long long)S * H);
    }
}
